// ASTNodesEmbedder_9895604650636
// MI455X (gfx1250) — matrix operand order repaired and checked by reading, not compiled, not run on hardware
//
#include <hip/hip_runtime.h>
#include <hip/hip_bf16.h>
#include <stdint.h>

// ---------------------------------------------------------------------------
// Types for CDNA5 WMMA (wave32): v_wmma_f32_16x16x32_bf16
// ---------------------------------------------------------------------------
typedef __attribute__((ext_vector_type(16))) __bf16 v16bf;
typedef __attribute__((ext_vector_type(8)))  float  v8f;

union BFrag {
    v16bf v;
    uint4 q[2];
};

// A-matrix 16x32 bf16 fragment from LDS, per ISA 7.12.2 layout:
//   lane L: row m = L%16, half = L/16
//   elements 0..7  (VGPR0..3) = K = kt + half*8      .. +7
//   elements 8..15 (VGPR4..7) = K = kt + 16 + half*8 .. +7
__device__ __forceinline__ v16bf load_a_frag_lds(const __bf16* lds, int stride,
                                                 int kt, int lane) {
    const int m    = lane & 15;
    const int half = lane >> 4;
    const __bf16* p = lds + m * stride + kt + half * 8;
    BFrag f;
    f.q[0] = *(const uint4*)(p);        // K = kt + half*8 .. +7
    f.q[1] = *(const uint4*)(p + 16);   // K = kt + 16 + half*8 .. +7
    return f.v;
}

// B-matrix 32x16 bf16 fragment from global W (row-major W[n][k], ld = K):
//   lane L: column n = n0 + L%16, half = L/16;
//   elements 0..7 = K = kt + half*8 .. +7, elements 8..15 = K = kt + 16 + half*8 .. +7.
__device__ __forceinline__ v16bf load_b_frag_glb(const __bf16* W, int ldw,
                                                 int n0, int kt, int lane) {
    const int col  = n0 + (lane & 15);
    const int half = lane >> 4;
    const __bf16* p = W + (size_t)col * ldw + kt + half * 8;
    BFrag f;
    f.q[0] = *(const uint4*)(p);        // K = kt + half*8 .. +7
    f.q[1] = *(const uint4*)(p + 16);   // K = kt + 16 + half*8 .. +7
    return f.v;
}

// ---------------------------------------------------------------------------
// Kernel 1: convert f32 weights -> bf16 workspace (done once per launch;
// tiny, and halves L2 weight traffic for every GEMM tile afterwards)
// ---------------------------------------------------------------------------
__global__ void cvt_f32_bf16(const float* __restrict__ src,
                             __bf16* __restrict__ dst, int n) {
    int i = blockIdx.x * 256 + threadIdx.x;
    if (i < n) dst[i] = (__bf16)src[i];
}

// ---------------------------------------------------------------------------
// Kernel 2: base gather  enc[i,:] = node_type_table[types[i],:]
// float4-vectorized: 64 lanes per node row.
// ---------------------------------------------------------------------------
__global__ __launch_bounds__(256) void gather_base(
    const float* __restrict__ table, const int* __restrict__ types,
    float* __restrict__ out, int n_nodes) {
    size_t idx   = (size_t)blockIdx.x * 256 + threadIdx.x;
    size_t total = (size_t)n_nodes * 64;
    if (idx >= total) return;
    size_t node = idx >> 6;
    int    d4   = (int)(idx & 63);
    const float4* src = (const float4*)(table + (size_t)types[node] * 256);
    float4*       dst = (float4*)(out + node * 256);
    dst[d4] = src[d4];
}

// ---------------------------------------------------------------------------
// Kernel 3: disjoint leaf update GEMM.
//   out[node_idx[r], :] = [new_src[new_idx[r]], table[types[node_idx[r]]]] @ W^T + b
// One workgroup (256 thr = 8 wave32) per 16-row M-tile; wave w owns columns
// [32w, 32w+32). Leaf counts are exact multiples of 16 -> no divergence,
// EXEC all-ones around WMMA as required.
// ---------------------------------------------------------------------------
template <int K1>
__global__ __launch_bounds__(256) void leaf_update_gemm(
    const float* __restrict__ new_src, const int* __restrict__ new_idx,
    const int* __restrict__ node_idx,
    const int* __restrict__ ast_types, const float* __restrict__ type_table,
    const __bf16* __restrict__ Wbf, const float* __restrict__ bias,
    float* __restrict__ out) {
    constexpr int K_TOTAL = K1 + 256;
    constexpr int STRIDE  = K_TOTAL + 8;   // +16B pad: rows land 4 banks apart
    __shared__ __bf16 smem[16 * STRIDE];

    const int tile = blockIdx.x;
    const int tid  = threadIdx.x;

    // Cooperative gather of the 16 concatenated rows into LDS (f32 -> bf16).
    for (int i = tid; i < 16 * K_TOTAL; i += 256) {
        int row = i / K_TOTAL;
        int k   = i - row * K_TOTAL;
        int g   = tile * 16 + row;
        float v;
        if (k < K1) {
            v = new_src[(size_t)new_idx[g] * K1 + k];
        } else {
            v = type_table[(size_t)ast_types[node_idx[g]] * 256 + (k - K1)];
        }
        smem[row * STRIDE + k] = (__bf16)v;
    }
    __syncthreads();

    const int lane = tid & 31;
    const int wave = tid >> 5;
    const int n0   = wave * 32;

    v8f c0 = {};
    v8f c1 = {};
    for (int kt = 0; kt < K_TOTAL; kt += 32) {
        v16bf a  = load_a_frag_lds(smem, STRIDE, kt, lane);
        v16bf b0 = load_b_frag_glb(Wbf, K_TOTAL, n0,      kt, lane);
        v16bf b1 = load_b_frag_glb(Wbf, K_TOTAL, n0 + 16, kt, lane);
        c0 = __builtin_amdgcn_wmma_f32_16x16x32_bf16(false, a, false, b0,
                                                     (short)0, c0, false, false);
        c1 = __builtin_amdgcn_wmma_f32_16x16x32_bf16(false, a, false, b1,
                                                     (short)0, c1, false, false);
    }

    // Epilogue: C layout (f32 16x16): lane L, VGPR r -> m = r + 8*(L/16),
    // n = L%16. Add bias, scatter row to out[node_idx[...]].
    const int   ncol  = lane & 15;
    const int   half  = lane >> 4;
    const float bias0 = bias[n0 + ncol];
    const float bias1 = bias[n0 + 16 + ncol];
#pragma unroll
    for (int r = 0; r < 8; ++r) {
        int    m    = r + 8 * half;
        size_t grow = (size_t)node_idx[tile * 16 + m];
        out[grow * 256 + n0 + ncol]      = c0[r] + bias0;
        out[grow * 256 + n0 + 16 + ncol] = c1[r] + bias1;
    }
}

// ---------------------------------------------------------------------------
// Launch
// ---------------------------------------------------------------------------
extern "C" void kernel_launch(void* const* d_in, const int* in_sizes, int n_in,
                              void* d_out, int out_size, void* d_ws, size_t ws_size,
                              hipStream_t stream) {
    const float* identifiers_encodings = (const float*)d_in[0];
    const float* node_type_table       = (const float*)d_in[1];
    const float* prim_table            = (const float*)d_in[2];
    const float* mod_table             = (const float*)d_in[3];
    const float* W_id                  = (const float*)d_in[4];
    const float* b_id                  = (const float*)d_in[5];
    const float* W_prim                = (const float*)d_in[6];
    const float* b_prim                = (const float*)d_in[7];
    const float* W_mod                 = (const float*)d_in[8];
    const float* b_mod                 = (const float*)d_in[9];
    const int*   ast_node_types        = (const int*)d_in[10];
    const int*   id_identifier_idx     = (const int*)d_in[11];
    const int*   id_node_idx           = (const int*)d_in[12];
    const int*   prim_types            = (const int*)d_in[13];
    const int*   prim_node_idx         = (const int*)d_in[14];
    const int*   mod_ids               = (const int*)d_in[15];
    const int*   mod_node_idx          = (const int*)d_in[16];
    float*       out                   = (float*)d_out;

    constexpr int N_NODES = 500000;
    constexpr int N_ID    = 150000;
    constexpr int N_PRIM  = 60000;
    constexpr int N_MOD   = 30000;

    // bf16 weight staging in workspace
    __bf16* wid_bf   = (__bf16*)d_ws;                 // 256*512
    __bf16* wprim_bf = wid_bf + 256 * 512;            // 256*320
    __bf16* wmod_bf  = wprim_bf + 256 * 320;          // 256*320
    (void)ws_size; (void)in_sizes; (void)n_in; (void)out_size;

    cvt_f32_bf16<<<(256 * 512 + 255) / 256, 256, 0, stream>>>(W_id,   wid_bf,   256 * 512);
    cvt_f32_bf16<<<(256 * 320 + 255) / 256, 256, 0, stream>>>(W_prim, wprim_bf, 256 * 320);
    cvt_f32_bf16<<<(256 * 320 + 255) / 256, 256, 0, stream>>>(W_mod,  wmod_bf,  256 * 320);

    // Base gather over all nodes (leaf rows get overwritten afterwards;
    // stream ordering guarantees gather completes first).
    long total4 = (long)N_NODES * 64;
    gather_base<<<(unsigned)((total4 + 255) / 256), 256, 0, stream>>>(
        node_type_table, ast_node_types, out, N_NODES);

    // Three disjoint leaf updates (WMMA bf16, f32 accumulate).
    leaf_update_gemm<256><<<N_ID / 16, 256, 0, stream>>>(
        identifiers_encodings, id_identifier_idx, id_node_idx,
        ast_node_types, node_type_table, wid_bf, b_id, out);
    leaf_update_gemm<64><<<N_PRIM / 16, 256, 0, stream>>>(
        prim_table, prim_types, prim_node_idx,
        ast_node_types, node_type_table, wprim_bf, b_prim, out);
    leaf_update_gemm<64><<<N_MOD / 16, 256, 0, stream>>>(
        mod_table, mod_ids, mod_node_idx,
        ast_node_types, node_type_table, wmod_bf, b_mod, out);
}
